// TFNO2D_24988119728190
// MI455X (gfx1250) — matrix operand order repaired and checked by reading, not compiled, not run on hardware
//
#include <hip/hip_runtime.h>
#include <hip/hip_bf16.h>
#include <math.h>

// ---------------------------------------------------------------------------
// TFNO2D step for gfx1250 (MI455X): all channel mixing + truncated-DFT done as
// f16 WMMA GEMMs (v_wmma_f32_16x16x32_f16), wave32.
// Each wave computes a 16x32 output strip: 1 A fragment, 2 B fragments,
// 2 WMMA issues per K-step. Contiguous-A path uses b128 vector loads.
// ---------------------------------------------------------------------------

typedef __attribute__((ext_vector_type(16))) _Float16 v16h;
typedef __attribute__((ext_vector_type(8)))  float    v8f;

#define Bsz   8
#define Tsz   4
#define Hsz   128
#define Wsz   256
#define HW    (Hsz * Wsz)
#define NPIX  (Bsz * HW)          // 262144
#define CW    128                 // WIDTH
#define MODES 64                  // MH == MW
#define RI    32
#define RO    32
#define RH    16
#define RW    16
#define RL    2

__device__ __forceinline__ float gelu_erf(float x) {
  return 0.5f * x * (1.0f + erff(x * 0.70710678118654752f));
}

// ---------------------------------------------------------------------------
// Generic batched / strided WMMA GEMM:
//   C[bi, m, n] = act( sum_k A[bi, m, k] * Bw[n, k] + bias[n] + Radd[bi, m, n] )
// A is f32 with element strides (sAm, sAk); batch base = (bi/bdiv)*sAb1 +
// (bi%bdiv)*sAb2 (same decomposition for C). Bw is f16 row-major [Npad, K],
// K a multiple of 32, rows padded to a multiple of 32 (zero rows beyond N).
// One wave computes one 16x32 strip; grid-stride over batch*Mtiles*N32tiles.
// Fragment layouts follow CDNA5 ISA 7.12.2 (16-bit A 16x32, B 32x16, f32 C/D).
// ---------------------------------------------------------------------------
__global__ __launch_bounds__(256) void k_wmma_gemm(
    const float* __restrict__ A, long long sAm, long long sAk,
    long long sAb1, long long sAb2,
    const _Float16* __restrict__ Bw,
    const float* __restrict__ bias,
    const float* __restrict__ Radd,
    float* __restrict__ C, long long sCm, long long sCn,
    long long sCb1, long long sCb2,
    int M, int N, int K, int nbatch, int bdiv, int act)
{
  const int lane   = threadIdx.x & 31;
  const int lane15 = lane & 15;
  const int laneHi = lane >> 4;
  const int wid    = blockIdx.x * (blockDim.x >> 5) + (threadIdx.x >> 5);
  const int nw     = gridDim.x * (blockDim.x >> 5);

  const int tmn = (M + 15) >> 4;
  const int tnn = (N + 31) >> 5;                       // 32-column strips
  const long long tiles = (long long)nbatch * tmn * tnn;

  for (long long t = wid; t < tiles; t += nw) {
    const int bi = (int)(t / (tmn * tnn));
    const int r  = (int)(t % (tmn * tnn));
    const int tm = r / tnn;
    const int tn = r % tnn;

    const long long baseA = (long long)(bi / bdiv) * sAb1 + (long long)(bi % bdiv) * sAb2;
    const long long baseC = (long long)(bi / bdiv) * sCb1 + (long long)(bi % bdiv) * sCb2;

    int rowA = tm * 16 + lane15; if (rowA >= M) rowA = M - 1;
    const int nb0 = tn * 32 + lane15;                  // B rows padded to x32
    const float*    Arow  = A  + baseA + (long long)rowA * sAm;
    const _Float16* Brow0 = Bw + (long long)nb0 * K;
    const _Float16* Brow1 = Brow0 + (long long)16 * K;

    v8f acc0 = {}, acc1 = {};
    if (sAk == 1) {
      // Contiguous A: each lane's fragment = two 8-float runs -> 4x b128 loads
      for (int k0 = 0; k0 < K; k0 += 32) {
        if (k0 + 32 < K)
          __builtin_prefetch(&Arow[k0 + 32], 0, 1);
        const float4* ap = (const float4*)(Arow + k0 + (laneHi << 3));
        const float4  a0 = ap[0], a1 = ap[1];          // K = base..base+7
        const float4  a2 = ap[4], a3 = ap[5];          // K = base+16..base+23
        v16h af, bf0, bf1;
        af[0] = (_Float16)a0.x; af[1] = (_Float16)a0.y;
        af[2] = (_Float16)a0.z; af[3] = (_Float16)a0.w;
        af[4] = (_Float16)a1.x; af[5] = (_Float16)a1.y;
        af[6] = (_Float16)a1.z; af[7] = (_Float16)a1.w;
        af[8]  = (_Float16)a2.x; af[9]  = (_Float16)a2.y;
        af[10] = (_Float16)a2.z; af[11] = (_Float16)a2.w;
        af[12] = (_Float16)a3.x; af[13] = (_Float16)a3.y;
        af[14] = (_Float16)a3.z; af[15] = (_Float16)a3.w;
        #pragma unroll
        for (int e = 0; e < 16; ++e) {
          bf0[e] = Brow0[k0 + (e & 7) + ((e >> 3) << 4) + (laneHi << 3)];
          bf1[e] = Brow1[k0 + (e & 7) + ((e >> 3) << 4) + (laneHi << 3)];
        }
        acc0 = __builtin_amdgcn_wmma_f32_16x16x32_f16(false, af, false, bf0,
                                                      (short)0, acc0, false, false);
        acc1 = __builtin_amdgcn_wmma_f32_16x16x32_f16(false, af, false, bf1,
                                                      (short)0, acc1, false, false);
      }
    } else {
      for (int k0 = 0; k0 < K; k0 += 32) {
        if (k0 + 32 < K)
          __builtin_prefetch(&Arow[(long long)(k0 + 32) * sAk], 0, 1);
        v16h af, bf0, bf1;
        // A 16x32 layout: lanes0-15 M=lane, K={0..7,16..23}; lanes16-31 K={8..15,24..31}
        #pragma unroll
        for (int e = 0; e < 16; ++e) {
          const int ke = (e & 7) + ((e >> 3) << 4) + (laneHi << 3);
          af[e] = (_Float16)Arow[(long long)(k0 + ke) * sAk];
        }
        #pragma unroll
        for (int e = 0; e < 16; ++e) {
          bf0[e] = Brow0[k0 + (e & 7) + ((e >> 3) << 4) + (laneHi << 3)];
          bf1[e] = Brow1[k0 + (e & 7) + ((e >> 3) << 4) + (laneHi << 3)];
        }
        acc0 = __builtin_amdgcn_wmma_f32_16x16x32_f16(false, af, false, bf0,
                                                      (short)0, acc0, false, false);
        acc1 = __builtin_amdgcn_wmma_f32_16x16x32_f16(false, af, false, bf1,
                                                      (short)0, acc1, false, false);
      }
    }

    // D 16x16 f32 layout: lanes0-15 N=lane, M=r; lanes16-31 N=lane-16, M=8+r
    const int m0 = tm * 16 + (laneHi << 3);
    #pragma unroll
    for (int half = 0; half < 2; ++half) {
      const int cn = tn * 32 + half * 16 + lane15;
      if (cn < N) {
        const float bv = bias ? bias[cn] : 0.0f;
        const v8f acc = half ? acc1 : acc0;
        #pragma unroll
        for (int rr = 0; rr < 8; ++rr) {
          const int m = m0 + rr;
          if (m < M) {
            const long long off = baseC + (long long)m * sCm + (long long)cn * sCn;
            float v = acc[rr] + bv;
            if (Radd) v += Radd[off];
            if (act)  v = gelu_erf(v);
            C[off] = v;
          }
        }
      }
    }
  }
}

// ---------------------------------------------------------------------------
// Encoding (x, t, forcing, sphere position) + lift conv1x1 (23 -> 128)
// ---------------------------------------------------------------------------
__global__ void k_encode_lift(const float* __restrict__ x,
                              const float* __restrict__ tg,
                              const float* __restrict__ lw,
                              const float* __restrict__ lb,
                              float* __restrict__ X)
{
  const int n = blockIdx.x * blockDim.x + threadIdx.x;
  if (n >= NPIX) return;
  const int b = n / HW, p = n % HW;
  const int h = p / Wsz, w = p % Wsz;
  const float PI = 3.14159265358979323846f;
  const float theta = -0.5f * PI + PI * (float)h / (float)(Hsz - 1);
  const float phi   = 2.0f * PI * (float)w / (float)Wsz;

  float f[23];
  #pragma unroll
  for (int t = 0; t < Tsz; ++t) {
    const float tv = tg[b * Tsz + t];
    f[t*5+0] = x[(((long long)(b*Tsz+t)*3 + 0) * HW) + p];
    f[t*5+1] = x[(((long long)(b*Tsz+t)*3 + 1) * HW) + p];
    f[t*5+2] = x[(((long long)(b*Tsz+t)*3 + 2) * HW) + p];
    f[t*5+3] = tv;
    const float phic = tv * 2.0f * PI;
    const float thc  = sinf(tv * (2.0f * PI / 365.0f)) * 0.4f;
    const float d    = theta - thc;
    f[t*5+4] = cosf(phi - phic) * expf(-(d * d) / ((0.5f*PI)*(0.5f*PI)));
  }
  const float ct = cosf(theta);
  f[20] = ct * cosf(phi);
  f[21] = ct * sinf(phi);
  f[22] = sinf(theta);

  float* out = X + (long long)n * CW;
  for (int o = 0; o < CW; ++o) {
    float s = lb[o];
    const float* wr = lw + o * 23;
    #pragma unroll
    for (int k = 0; k < 23; ++k) s = fmaf(f[k], wr[k], s);
    out[o] = s;
  }
}

// ---------------------------------------------------------------------------
// Channel LayerNorm over 128 contiguous channels; one wave per row.
// ---------------------------------------------------------------------------
__global__ void k_chan_ln(const float* __restrict__ src, float* __restrict__ dst,
                          const float* __restrict__ g, const float* __restrict__ bta,
                          long long rows)
{
  const int lane = threadIdx.x & 31;
  const long long row = (long long)blockIdx.x * (blockDim.x >> 5) + (threadIdx.x >> 5);
  if (row >= rows) return;
  const float* s = src + row * CW;
  float v[4], sum = 0.f, sq = 0.f;
  #pragma unroll
  for (int i = 0; i < 4; ++i) { v[i] = s[lane*4 + i]; sum += v[i]; sq += v[i]*v[i]; }
  #pragma unroll
  for (int m = 16; m >= 1; m >>= 1) { sum += __shfl_xor(sum, m, 32); sq += __shfl_xor(sq, m, 32); }
  const float mu  = sum * (1.0f / CW);
  const float var = sq  * (1.0f / CW) - mu * mu;
  const float inv = rsqrtf(var + 1e-5f);
  float* d = dst + row * CW;
  #pragma unroll
  for (int i = 0; i < 4; ++i) { const int c = lane*4 + i; d[c] = (v[i]-mu)*inv*g[c] + bta[c]; }
}

// ---------------------------------------------------------------------------
// Weight conversion / packing
// ---------------------------------------------------------------------------
__global__ void k_f32_to_f16(const float* __restrict__ s, _Float16* __restrict__ d, int n) {
  const int i = blockIdx.x * blockDim.x + threadIdx.x;
  if (i < n) d[i] = (_Float16)s[i];
}
__global__ void k_pack_p2(const float* __restrict__ s, _Float16* __restrict__ d) {
  const int i = blockIdx.x * blockDim.x + threadIdx.x;   // 32 x 64 (rows 3..31 zero)
  if (i >= 32 * 64) return;
  const int r = i >> 6, c = i & 63;
  d[i] = (_Float16)((r < 3) ? s[r * 64 + c] : 0.0f);
}
__global__ void k_pack_UI(const float* __restrict__ s, _Float16* __restrict__ d) {
  // s: U_I [128 c][32 i'] complex-interleaved -> d [64 n][128 c]; n<32:Re, n>=32:Im
  const int i = blockIdx.x * blockDim.x + threadIdx.x;   // 64*128
  if (i >= 64 * 128) return;
  const int nrow = i >> 7, c = i & 127;
  d[i] = (_Float16)s[(c * RI + (nrow & 31)) * 2 + (nrow >> 5)];
}
__global__ void k_pack_UO(const float* __restrict__ s, _Float16* __restrict__ d) {
  // s: U_O [128 o][32 o'] complex -> d [128 o][64 k]; k<32: Re, k>=32: -Im
  const int i = blockIdx.x * blockDim.x + threadIdx.x;   // 128*64
  if (i >= 128 * 64) return;
  const int o = i >> 6, k = i & 63, im = k >> 5;
  const float v = s[(o * RO + (k & 31)) * 2 + im];
  d[i] = (_Float16)(im ? -v : v);
}

// DFT tables (orthonormal; inverse-W folds hermitian a_q weights + Re/Im split)
__global__ void k_dft_tables(_Float16* __restrict__ Dw,  _Float16* __restrict__ Dh,
                             _Float16* __restrict__ Dhic,_Float16* __restrict__ Dhis,
                             _Float16* __restrict__ DwiRe,_Float16* __restrict__ DwiIm)
{
  const float PI2 = 6.28318530717958647692f;
  const float SW = 1.0f / 16.0f;            // 1/sqrt(256)
  const float SH = 0.08838834764831845f;    // 1/sqrt(128)
  const int i = blockIdx.x * blockDim.x + threadIdx.x;
  if (i < 128 * 256) {  // Dw [n=64q x {cos,-sin}][w]
    const int nn = i >> 8, w = i & 255, q = nn & 63;
    const float a = PI2 * (float)((w * q) & 255) / 256.0f;
    Dw[i] = (_Float16)(((nn < 64) ? cosf(a) : -sinf(a)) * SW);
  }
  if (i < 128 * 128) {  // Dh [n=64p x {cos,-sin}][h]
    const int nn = i >> 7, h = i & 127, q = nn & 63;
    const float a = PI2 * (float)((h * q) & 127) / 128.0f;
    Dh[i] = (_Float16)(((nn < 64) ? cosf(a) : -sinf(a)) * SH);
  }
  if (i < 128 * 64) {   // Dhic/Dhis [h][p]: +cos / +sin (e^{+i})
    const int h = i >> 6, p = i & 63;
    const float a = PI2 * (float)((h * p) & 127) / 128.0f;
    Dhic[i] = (_Float16)(cosf(a) * SH);
    Dhis[i] = (_Float16)(sinf(a) * SH);
  }
  if (i < 256 * 128) {  // DwiRe/DwiIm [w][k]: k<64 re-plane coeff, k>=64 im-plane
    const int w = i >> 7, k = i & 127, q = k & 63;
    const float aq = (q == 0) ? 1.0f : 2.0f;
    const float a  = PI2 * (float)((w * q) & 255) / 256.0f;
    DwiRe[i] = (_Float16)(((k < 64) ?  cosf(a) : -sinf(a)) * aq * SW);
    DwiIm[i] = (_Float16)(((k < 64) ?  sinf(a) :  cosf(a)) * aq * SW);
  }
}

// ---------------------------------------------------------------------------
// Tucker core contractions
// ---------------------------------------------------------------------------
__global__ void k_gl(const float* __restrict__ G, const float* __restrict__ uL,
                     float* __restrict__ Gl)
{
  // Gl[o',i',rp,rq] = sum_l uL[l] * G[l,o',i',rp,rq]   (complex)
  const int i = blockIdx.x * blockDim.x + threadIdx.x;  // 32*32*16*16
  if (i >= RO * RI * RH * RW) return;
  float re = 0.f, im = 0.f;
  #pragma unroll
  for (int l = 0; l < RL; ++l) {
    const float ur = uL[l*2], ui = uL[l*2+1];
    const float gr = G[((long long)l * (RO*RI*RH*RW) + i) * 2];
    const float gi = G[((long long)l * (RO*RI*RH*RW) + i) * 2 + 1];
    re += ur * gr - ui * gi;
    im += ur * gi + ui * gr;
  }
  Gl[i*2] = re; Gl[i*2+1] = im;
}

__global__ void k_wm(const float* __restrict__ Gl, const float* __restrict__ UH,
                     const float* __restrict__ UW, float* __restrict__ Wm)
{
  // Wm[mode=p*64+q][o'][i'] = sum_{rp,rq} UH[p,rp]*UW[q,rq]*Gl[o',i',rp,rq]
  const long long i = (long long)blockIdx.x * blockDim.x + threadIdx.x;
  if (i >= (long long)MODES * MODES * RO * RI) return;
  const int ii = (int)(i & 31);
  const int oo = (int)((i >> 5) & 31);
  const int q  = (int)((i >> 10) & 63);
  const int p  = (int)(i >> 16);
  const float* g = Gl + (long long)(oo * RI + ii) * (RH * RW) * 2;
  float re = 0.f, im = 0.f;
  for (int rp = 0; rp < RH; ++rp) {
    const float hr = UH[(p*RH + rp)*2], hi = UH[(p*RH + rp)*2 + 1];
    #pragma unroll
    for (int rq = 0; rq < RW; ++rq) {
      const float wr = UW[(q*RW + rq)*2], wi = UW[(q*RW + rq)*2 + 1];
      const float tr = hr*wr - hi*wi;
      const float ti = hr*wi + hi*wr;
      const float gr = g[(rp*RW + rq)*2], gi = g[(rp*RW + rq)*2 + 1];
      re += tr*gr - ti*gi;
      im += tr*gi + ti*gr;
    }
  }
  const long long mo = (((long long)(p*64 + q)) * RO + oo) * RI + ii;
  Wm[mo*2] = re; Wm[mo*2+1] = im;
}

// Sm[b,mode,o'] = sum_i' Wm[mode,o',i'] * Fm[b,mode,i']   (complex, planar 32re+32im)
__global__ void k_modemix(const float* __restrict__ Wm, const float* __restrict__ Fm,
                          float* __restrict__ Sm)
{
  const int i = blockIdx.x * blockDim.x + threadIdx.x;    // 8*4096*32
  if (i >= Bsz * 4096 * RO) return;
  const int oo = i & 31;
  const int mode = (i >> 5) & 4095;
  const int b = i >> 17;
  const float* wrow = Wm + (((long long)mode * RO + oo) * RI) * 2;
  const float* f = Fm + (long long)(b * 4096 + mode) * 64;
  float re = 0.f, im = 0.f;
  #pragma unroll 8
  for (int ii = 0; ii < RI; ++ii) {
    const float wr = wrow[ii*2], wi = wrow[ii*2+1];
    const float fr = f[ii], fi = f[32 + ii];
    re += wr*fr - wi*fi;
    im += wr*fi + wi*fr;
  }
  float* s = Sm + (long long)(b * 4096 + mode) * 64;
  s[oo] = re; s[32 + oo] = im;
}

// ---------------------------------------------------------------------------
// Complex recombines after cos/(-sin) GEMMs
// ---------------------------------------------------------------------------
__global__ void k_recomb_w(const float* __restrict__ Rw, float* __restrict__ Fw)
{
  const int i = blockIdx.x * blockDim.x + threadIdx.x;    // 1024*64*32
  if (i >= 1024 * 64 * 32) return;
  const int ii = i & 31, q = (i >> 5) & 63, bh = i >> 11;
  const float* r = Rw + (long long)bh * 8192;             // [ch64][col128]
  const float fre = r[ii*128 + q]        - r[(ii+32)*128 + 64 + q];
  const float fim = r[(ii+32)*128 + q]   + r[ii*128 + 64 + q];
  float* o = Fw + ((long long)bh * 64 + q) * 64;
  o[ii] = fre; o[32 + ii] = fim;
}
__global__ void k_recomb_h(const float* __restrict__ Rh, float* __restrict__ Fm)
{
  const int i = blockIdx.x * blockDim.x + threadIdx.x;    // 8*64*64*32
  if (i >= Bsz * 64 * 64 * 32) return;
  const int ii = i & 31, q = (i >> 5) & 63, p = (i >> 11) & 63, b = i >> 17;
  const float* r = Rh + (long long)(b * 64 + q) * 8192;   // [ch64][col128]
  const float fre = r[ii*128 + p]        - r[(ii+32)*128 + 64 + p];
  const float fim = r[(ii+32)*128 + p]   + r[ii*128 + 64 + p];
  float* o = Fm + (long long)(b * 4096 + p * 64 + q) * 64;
  o[ii] = fre; o[32 + ii] = fim;
}
__global__ void k_recomb_hi(const float* __restrict__ Rhc, const float* __restrict__ Rhs,
                            float* __restrict__ Gm)
{
  const int i = blockIdx.x * blockDim.x + threadIdx.x;    // 8*128*64*32
  if (i >= Bsz * Hsz * 64 * 32) return;
  const int oo = i & 31, q = (i >> 5) & 63, h = (i >> 11) & 127, b = i >> 18;
  const long long bq = (long long)(b * 64 + q) * 8192;    // [ch64][h128]
  const float gre = Rhc[bq + oo*128 + h]        - Rhs[bq + (oo+32)*128 + h];
  const float gim = Rhc[bq + (oo+32)*128 + h]   + Rhs[bq + oo*128 + h];
  const long long base = (long long)(b * Hsz + h) * 128;  // [k=part*64+q][o'32]
  Gm[(base + q) * 32 + oo]      = gre;
  Gm[(base + 64 + q) * 32 + oo] = gim;
}

// ---------------------------------------------------------------------------
// Host side
// ---------------------------------------------------------------------------
static void run_gemm(hipStream_t s,
                     const float* A, long long sAm, long long sAk,
                     long long sAb1, long long sAb2,
                     const _Float16* Bw, const float* bias, const float* Radd,
                     float* C, long long sCm, long long sCn,
                     long long sCb1, long long sCb2,
                     int M, int N, int K, int nbatch, int bdiv, int act)
{
  const long long tiles = (long long)nbatch * ((M + 15) / 16) * ((N + 31) / 32);
  long long blocks = (tiles + 7) / 8;
  if (blocks > 32768) blocks = 32768;
  if (blocks < 1)     blocks = 1;
  k_wmma_gemm<<<(int)blocks, 256, 0, s>>>(A, sAm, sAk, sAb1, sAb2, Bw, bias, Radd,
                                          C, sCm, sCn, sCb1, sCb2,
                                          M, N, K, nbatch, bdiv, act);
}

extern "C" void kernel_launch(void* const* d_in, const int* in_sizes, int n_in,
                              void* d_out, int out_size, void* d_ws, size_t ws_size,
                              hipStream_t stream)
{
  (void)in_sizes; (void)n_in; (void)out_size; (void)ws_size;

  const float* in_x      = (const float*)d_in[0];
  const float* in_tgrid  = (const float*)d_in[1];
  const float* in_liftw  = (const float*)d_in[2];
  const float* in_liftb  = (const float*)d_in[3];
  const float* in_ln1g   = (const float*)d_in[4];
  const float* in_ln1b   = (const float*)d_in[5];
  const float* in_skipw  = (const float*)d_in[6];
  const float* in_skipb  = (const float*)d_in[7];
  const float* in_ln2g   = (const float*)d_in[8];
  const float* in_ln2b   = (const float*)d_in[9];
  const float* in_fc1w   = (const float*)d_in[10];
  const float* in_fc1b   = (const float*)d_in[11];
  const float* in_fc2w   = (const float*)d_in[12];
  const float* in_fc2b   = (const float*)d_in[13];
  const float* in_G      = (const float*)d_in[14];   // complex interleaved
  const float* in_UL     = (const float*)d_in[15];
  const float* in_UO     = (const float*)d_in[16];
  const float* in_UI     = (const float*)d_in[17];
  const float* in_UH     = (const float*)d_in[18];
  const float* in_UW     = (const float*)d_in[19];
  const float* in_p1w    = (const float*)d_in[20];
  const float* in_p1b    = (const float*)d_in[21];
  const float* in_p2w    = (const float*)d_in[22];
  const float* in_p2b    = (const float*)d_in[23];
  float* out = (float*)d_out;

  // ---- workspace carve-up -------------------------------------------------
  char* wsb = (char*)d_ws;
  size_t off = 0;
  auto alloc = [&](size_t bytes) -> void* {
    void* p = wsb + off;
    off = (off + bytes + 255) & ~(size_t)255;
    return p;
  };
  // f16 weight / table area (all row counts padded to multiples of 32)
  _Float16* h_skip  = (_Float16*)alloc(sizeof(_Float16) * 4 * 128 * 128);
  _Float16* h_fc1   = (_Float16*)alloc(sizeof(_Float16) * 4 * 64 * 128);
  _Float16* h_fc2   = (_Float16*)alloc(sizeof(_Float16) * 4 * 128 * 64);
  _Float16* h_p1    = (_Float16*)alloc(sizeof(_Float16) * 64 * 128);
  _Float16* h_p2    = (_Float16*)alloc(sizeof(_Float16) * 32 * 64);
  _Float16* h_UI    = (_Float16*)alloc(sizeof(_Float16) * 64 * 128);
  _Float16* h_UO    = (_Float16*)alloc(sizeof(_Float16) * 128 * 64);
  _Float16* h_Dw    = (_Float16*)alloc(sizeof(_Float16) * 128 * 256);
  _Float16* h_Dh    = (_Float16*)alloc(sizeof(_Float16) * 128 * 128);
  _Float16* h_Dhic  = (_Float16*)alloc(sizeof(_Float16) * 128 * 64);
  _Float16* h_Dhis  = (_Float16*)alloc(sizeof(_Float16) * 128 * 64);
  _Float16* h_DwiRe = (_Float16*)alloc(sizeof(_Float16) * 256 * 128);
  _Float16* h_DwiIm = (_Float16*)alloc(sizeof(_Float16) * 256 * 128);
  // f32 activation area
  float* X   = (float*)alloc(sizeof(float) * (size_t)NPIX * 128);
  float* T1  = (float*)alloc(sizeof(float) * (size_t)NPIX * 128);
  float* T2  = (float*)alloc(sizeof(float) * (size_t)NPIX * 128);
  float* Z64 = (float*)alloc(sizeof(float) * (size_t)NPIX * 64);   // UI-out / invW-out / fc1-out / p1-out
  float* Rw  = (float*)alloc(sizeof(float) * (size_t)1024 * 8192); // also aliased: Rhc
  float* Fw  = (float*)alloc(sizeof(float) * (size_t)1024 * 64 * 64); // aliased: Rhs
  float* Rh  = (float*)alloc(sizeof(float) * (size_t)512 * 8192);  // aliased: Gm
  float* Fm  = (float*)alloc(sizeof(float) * (size_t)Bsz * 4096 * 64);
  float* Sm  = (float*)alloc(sizeof(float) * (size_t)Bsz * 4096 * 64);
  float* Gl  = (float*)alloc(sizeof(float) * (size_t)RO * RI * RH * RW * 2);
  float* Wm  = (float*)alloc(sizeof(float) * (size_t)4096 * RO * RI * 2);
  float* Rhc = Rw;   // safe: Rw dead after k_recomb_w
  float* Rhs = Fw;   // safe: Fw dead after H-DFT
  float* Gm  = Rh;   // safe: Rh dead after k_recomb_h

  const int EW = 256;
  // ---- one-time per call: weight conversion + DFT tables ------------------
  k_f32_to_f16<<<(4*128*128 + EW-1)/EW, EW, 0, stream>>>(in_skipw, h_skip, 4*128*128);
  k_f32_to_f16<<<(4*64*128  + EW-1)/EW, EW, 0, stream>>>(in_fc1w,  h_fc1,  4*64*128);
  k_f32_to_f16<<<(4*128*64  + EW-1)/EW, EW, 0, stream>>>(in_fc2w,  h_fc2,  4*128*64);
  k_f32_to_f16<<<(64*128    + EW-1)/EW, EW, 0, stream>>>(in_p1w,   h_p1,   64*128);
  k_pack_p2<<<(32*64  + EW-1)/EW, EW, 0, stream>>>(in_p2w, h_p2);
  k_pack_UI<<<(64*128 + EW-1)/EW, EW, 0, stream>>>(in_UI, h_UI);
  k_pack_UO<<<(128*64 + EW-1)/EW, EW, 0, stream>>>(in_UO, h_UO);
  k_dft_tables<<<(128*256 + EW-1)/EW, EW, 0, stream>>>(h_Dw, h_Dh, h_Dhic, h_Dhis, h_DwiRe, h_DwiIm);

  // ---- encode + lift ------------------------------------------------------
  k_encode_lift<<<(NPIX + EW-1)/EW, EW, 0, stream>>>(in_x, in_tgrid, in_liftw, in_liftb, X);

  // ---- 4 TFNO blocks ------------------------------------------------------
  for (int l = 0; l < 4; ++l) {
    // Tucker core for this layer
    k_gl<<<(RO*RI*RH*RW + EW-1)/EW, EW, 0, stream>>>(in_G, in_UL + l*RL*2, Gl);
    k_wm<<<(4096*RO*RI + EW-1)/EW, EW, 0, stream>>>(Gl, in_UH, in_UW, Wm);

    // y = LN1(out)
    k_chan_ln<<<(NPIX + 7)/8, 256, 0, stream>>>(X, T1, in_ln1g + l*128, in_ln1b + l*128, NPIX);

    // Z = y @ U_I  (complex -> 32re+32im channels)
    run_gemm(stream, T1, 128, 1, 0, 0, h_UI, nullptr, nullptr,
             Z64, 64, 1, 0, 0, NPIX, 64, 128, 1, 1, 0);

    // W-DFT: batch (b,h); A[ch64, w256]; N = 64q x {cos,-sin}
    run_gemm(stream, Z64, 1, 64, 16384, 0, h_Dw, nullptr, nullptr,
             Rw, 128, 1, 8192, 0, 64, 128, 256, 1024, 1, 0);
    k_recomb_w<<<(1024*64*32 + EW-1)/EW, EW, 0, stream>>>(Rw, Fw);

    // H-DFT: batch (b,q) with bdiv=64; A[ch64, h128]
    run_gemm(stream, Fw, 1, 4096, 524288, 64, h_Dh, nullptr, nullptr,
             Rh, 128, 1, 524288, 8192, 64, 128, 128, 512, 64, 0);
    k_recomb_h<<<(Bsz*64*64*32 + EW-1)/EW, EW, 0, stream>>>(Rh, Fm);

    // per-mode rank-space complex mixing
    k_modemix<<<(Bsz*4096*32 + EW-1)/EW, EW, 0, stream>>>(Wm, Fm, Sm);

    // inverse H-DFT (e^{+i}): cos and sin halves
    run_gemm(stream, Sm, 1, 4096, 262144, 64, h_Dhic, nullptr, nullptr,
             Rhc, 128, 1, 524288, 8192, 64, 128, 64, 512, 64, 0);
    run_gemm(stream, Sm, 1, 4096, 262144, 64, h_Dhis, nullptr, nullptr,
             Rhs, 128, 1, 524288, 8192, 64, 128, 64, 512, 64, 0);
    k_recomb_hi<<<(Bsz*Hsz*64*32 + EW-1)/EW, EW, 0, stream>>>(Rhc, Rhs, Gm);

    // inverse W-DFT with hermitian a_q weights -> complex spatial S (Re/Im planes)
    run_gemm(stream, Gm, 1, 32, 4096, 0, h_DwiRe, nullptr, nullptr,
             Z64, 1, 64, 16384, 0, 32, 256, 128, 1024, 1, 0);
    run_gemm(stream, Gm, 1, 32, 4096, 0, h_DwiIm, nullptr, nullptr,
             Z64 + 32, 1, 64, 16384, 0, 32, 256, 128, 1024, 1, 0);

    // skip conv1x1 (with bias)
    run_gemm(stream, X, 128, 1, 0, 0, h_skip + (size_t)l*128*128,
             in_skipb + l*128, nullptr, T2, 128, 1, 0, 0, NPIX, 128, 128, 1, 1, 0);

    // y = gelu( Re{S @ U_O} + skip )
    run_gemm(stream, Z64, 64, 1, 0, 0, h_UO, nullptr, T2,
             T1, 128, 1, 0, 0, NPIX, 128, 64, 1, 1, 1);

    // MLP branch: z = fc2(gelu(fc1(LN2(y)))); out = gelu(y + z)
    k_chan_ln<<<(NPIX + 7)/8, 256, 0, stream>>>(T1, T2, in_ln2g + l*128, in_ln2b + l*128, NPIX);
    run_gemm(stream, T2, 128, 1, 0, 0, h_fc1 + (size_t)l*64*128,
             in_fc1b + l*64, nullptr, Z64, 64, 1, 0, 0, NPIX, 64, 128, 1, 1, 1);
    run_gemm(stream, Z64, 64, 1, 0, 0, h_fc2 + (size_t)l*128*64,
             in_fc2b + l*128, T1, X, 128, 1, 0, 0, NPIX, 128, 64, 1, 1, 1);
  }

  // ---- projection head ----------------------------------------------------
  run_gemm(stream, X, 128, 1, 0, 0, h_p1, in_p1b, nullptr,
           Z64, 64, 1, 0, 0, NPIX, 64, 128, 1, 1, 1);
  // p2: C layout [b][o][h][w]; batch = b, rows = pixel-in-image, cols = o
  run_gemm(stream, Z64, 64, 1, (long long)HW * 64, 0, h_p2, in_p2b, nullptr,
           out, 1, HW, (long long)3 * HW, 0, HW, 3, 64, Bsz, 1, 0);
}
